// RecurrentSNN_89292370084011
// MI455X (gfx1250) — matrix operand order repaired and checked by reading, not compiled, not run on hardware
//
#include <hip/hip_runtime.h>

// RecurrentSNN forward on gfx1250 (wave32, WMMA f16, async-LDS staging):
//   cur1 = x @ W_in^T + b_in   -> leaky integrate -> spike
//   cur_out = spk1 @ W_out^T + b_out -> leaky integrate -> spike
// B=256 T=1024 F=64 H=512 O=64. One workgroup per 16-batch-row chain (16 WGs),
// 8 waves, all weights pinned in VGPRs as f16 fragments, x tiles double-buffered
// in LDS via global_load_async_to_lds_b128 (ASYNCcnt), spikes staged in LDS.

typedef _Float16 v16h __attribute__((ext_vector_type(16)));
typedef float    v8f  __attribute__((ext_vector_type(8)));

#define SNN_B 256
#define SNN_T 1024
#define SNN_F 64
#define SNN_H 512
#define SNN_O 64
#define SNN_BETA   0.9f
#define SNN_THRESH 1.0f

// ---- CDNA5 WMMA fragment loaders (wave32 layouts per CDNA5 ISA) ----

// A fragment (16x32, 16-bit): lane L -> row M=L&15;
// halves 0..7  = K[(L<16?0:8)  + kbase .. +7]
// halves 8..15 = K[(L<16?16:24)+ kbase .. +7]
__device__ __forceinline__ v16h a_frag_from_lds_f32(const float* tile, int ldr,
                                                    int kbase, int lane) {
  const int m   = lane & 15;
  const int klo = kbase + ((lane < 16) ? 0 : 8);
  const int khi = kbase + ((lane < 16) ? 16 : 24);
  const float* row = tile + m * ldr;
  v16h a;
#pragma unroll
  for (int i = 0; i < 8; ++i) a[i] = (_Float16)row[klo + i];
#pragma unroll
  for (int i = 0; i < 8; ++i) a[8 + i] = (_Float16)row[khi + i];
  return a;
}

__device__ __forceinline__ v16h a_frag_from_lds_f16(const _Float16* spk, int kbase, int lane) {
  const int m   = lane & 15;
  const int klo = kbase + ((lane < 16) ? 0 : 8);
  const int khi = kbase + ((lane < 16) ? 16 : 24);
  const _Float16* row = spk + m * SNN_H;
  v16h a;
#pragma unroll
  for (int i = 0; i < 8; ++i) a[i] = row[klo + i];
#pragma unroll
  for (int i = 0; i < 8; ++i) a[8 + i] = row[khi + i];
  return a;
}

// B fragment (32x16, 16-bit): lane L -> col N=L&15;
// halves 0..7  = K[(L<16?0:8)  + kbase .. +7]
// halves 8..15 = K[(L<16?16:24)+ kbase .. +7]
// Source is W[n][k] row-major (W_in[H][F] / W_out[O][H]) i.e. B = W^T.
__device__ __forceinline__ v16h b_frag_from_wT(const float* W, int ldw,
                                               int n0, int kbase, int lane) {
  const int n  = n0 + (lane & 15);
  const int k0 = kbase + ((lane < 16) ? 0 : 8);
  const float* src = W + n * ldw + k0;
  v16h b;
#pragma unroll
  for (int i = 0; i < 16; ++i) b[i] = (_Float16)src[(i < 8) ? i : (i + 8)];
  return b;
}

// Async-copy one 16x64 f32 x tile (4 KB) into LDS: 256 threads x 16 B.
// GVS addressing: SGPR pair = x base, per-lane VGPR = 32-bit byte offset.
__device__ __forceinline__ void async_load_x_tile(const float* __restrict__ x,
                                                  unsigned lds_dst, int b0, int t,
                                                  int tid) {
  const int r = tid >> 4;   // batch row within tile
  const int c = tid & 15;   // 16-byte chunk within row
  const unsigned goff =
      (unsigned)(((b0 + r) * (SNN_T * SNN_F) + t * SNN_F) * 4 + c * 16);
  const unsigned loff = lds_dst + (unsigned)(r * (SNN_F * 4) + c * 16);
  asm volatile("global_load_async_to_lds_b128 %0, %1, %2 offset:0"
               :
               : "v"(loff), "v"(goff), "s"(x)
               : "memory");
}

// C/D 16x16 f32 layout: lane L, reg r -> (M = r + 8*(L>>4), N = L&15)

__global__ __launch_bounds__(256) void snn_scan_kernel(
    const float* __restrict__ x, const float* __restrict__ W_in,
    const float* __restrict__ b_in, const float* __restrict__ W_out,
    const float* __restrict__ b_out,
    float* __restrict__ mem_rec, float* __restrict__ spk_rec) {
  __shared__ _Float16 spk_lds[16 * SNN_H];            // 16 KB: spk1 [16][512]
  __shared__ float    red_lds[4 * 32 * 8];            //  4 KB: GEMM2 partials
  __shared__ __align__(16) float x_lds[2][16 * SNN_F]; //  8 KB: x double buffer

  const int tid    = threadIdx.x;
  const int lane   = tid & 31;
  const int w      = tid >> 5;        // wave 0..7
  const int b0     = blockIdx.x * 16; // batch tile base
  const int ncol   = lane & 15;       // N index inside a 16-wide tile
  const int rowoff = 8 * (lane >> 4); // C/D row offset

  // LDS byte offsets of the two x buffers (flat addr low 32 bits == LDS offset)
  const unsigned xbuf[2] = {(unsigned)(uintptr_t)&x_lds[0][0],
                            (unsigned)(uintptr_t)&x_lds[1][0]};

  // ---- one-time setup: pin f16 weight fragments + biases in registers ----
  v16h Bin[4][2];                     // W_in^T for this wave's 64 H-columns
  float bi[4];
#pragma unroll
  for (int j = 0; j < 4; ++j) {
    const int h0 = 64 * w + 16 * j;
    Bin[j][0] = b_frag_from_wT(W_in, SNN_F, h0, 0, lane);
    Bin[j][1] = b_frag_from_wT(W_in, SNN_F, h0, 32, lane);
    bi[j]     = b_in[h0 + ncol];
  }
  const int nt = w & 3;               // O-tile (16 cols) this wave contributes to
  const int kh = w >> 2;              // which K half of H=512
  v16h Bout[8];                       // W_out^T, K slice [kh*256, kh*256+256)
#pragma unroll
  for (int kf = 0; kf < 8; ++kf)
    Bout[kf] = b_frag_from_wT(W_out, SNN_H, 16 * nt, kh * 256 + kf * 32, lane);
  const float bo = b_out[16 * nt + ncol];

  v8f mem1[4] = {};                   // hidden membrane (this wave's 64 cols)
  v8f memo    = {};                   // output membrane (waves 0..3, tile nt)

  // Prologue: stage x tile for t=0
  async_load_x_tile(x, xbuf[0], b0, 0, tid);

#pragma unroll 1
  for (int t = 0; t < SNN_T; ++t) {
    // ---- prefetch t+1, then wait for tile t (async done-tokens are in-order)
    if (t + 1 < SNN_T) {
      async_load_x_tile(x, xbuf[(t + 1) & 1], b0, t + 1, tid);
      asm volatile("s_wait_asynccnt 0x1" ::: "memory");
    } else {
      asm volatile("s_wait_asynccnt 0x0" ::: "memory");
    }
    __syncthreads();  // B0: tile t visible to all waves

    // ---------- layer 1: cur1 = x_t @ W_in^T + b_in ; integrate ; spike ----
    const float* xt = &x_lds[t & 1][0];
    const v16h a0 = a_frag_from_lds_f32(xt, SNN_F, 0, lane);
    const v16h a1 = a_frag_from_lds_f32(xt, SNN_F, 32, lane);

#pragma unroll
    for (int j = 0; j < 4; ++j) {
      v8f c = {};
      c = __builtin_amdgcn_wmma_f32_16x16x32_f16(false, a0, false, Bin[j][0],
                                                 (short)0, c, false, false);
      c = __builtin_amdgcn_wmma_f32_16x16x32_f16(false, a1, false, Bin[j][1],
                                                 (short)0, c, false, false);
      const int hcol = 64 * w + 16 * j + ncol;
#pragma unroll
      for (int r = 0; r < 8; ++r) {
        const float cur  = c[r] + bi[j];
        const float old  = mem1[j][r];
        const float mnew = SNN_BETA * old + cur
                         - ((old > SNN_THRESH) ? SNN_THRESH : 0.0f);
        mem1[j][r] = mnew;
        spk_lds[(r + rowoff) * SNN_H + hcol] =
            (_Float16)((mnew > SNN_THRESH) ? 1.0f : 0.0f);
      }
    }

    __syncthreads();  // B1: spikes visible to all waves

    // ---------- layer 2: cur_out = spk1 @ W_out^T + b_out (K-split) -------
    v8f c2 = {};
#pragma unroll
    for (int kf = 0; kf < 8; ++kf) {
      const v16h a = a_frag_from_lds_f16(spk_lds, kh * 256 + kf * 32, lane);
      c2 = __builtin_amdgcn_wmma_f32_16x16x32_f16(false, a, false, Bout[kf],
                                                  (short)0, c2, false, false);
    }
    if (w >= 4) {
      float* slot = &red_lds[((w - 4) * 32 + lane) * 8];
#pragma unroll
      for (int r = 0; r < 8; ++r) slot[r] = c2[r];
    }

    __syncthreads();  // B2: partials ready; fences spk_lds/x_lds reuse

    if (w < 4) {
      const float* slot = &red_lds[(w * 32 + lane) * 8];
#pragma unroll
      for (int r = 0; r < 8; ++r) {
        const float cur  = c2[r] + slot[r] + bo;
        const float old  = memo[r];
        const float mnew = SNN_BETA * old + cur
                         - ((old > SNN_THRESH) ? SNN_THRESH : 0.0f);
        memo[r] = mnew;
        const float sp = (mnew > SNN_THRESH) ? 1.0f : 0.0f;
        const int brow = b0 + rowoff + r;
        const int idx  = (brow * SNN_T + t) * SNN_O + 16 * nt + ncol;
        mem_rec[idx] = mnew;
        spk_rec[idx] = sp;
      }
    }
  }
}

extern "C" void kernel_launch(void* const* d_in, const int* in_sizes, int n_in,
                              void* d_out, int out_size, void* d_ws, size_t ws_size,
                              hipStream_t stream) {
  (void)in_sizes; (void)n_in; (void)out_size; (void)d_ws; (void)ws_size;
  const float* x     = (const float*)d_in[0];
  const float* W_in  = (const float*)d_in[1];
  const float* b_in  = (const float*)d_in[2];
  const float* W_out = (const float*)d_in[3];
  const float* b_out = (const float*)d_in[4];
  float* mem_rec = (float*)d_out;
  float* spk_rec = mem_rec + (size_t)SNN_B * SNN_T * SNN_O;
  snn_scan_kernel<<<SNN_B / 16, 256, 0, stream>>>(x, W_in, b_in, W_out, b_out,
                                                  mem_rec, spk_rec);
}
